// SpatialCrossAttention_29626684408001
// MI455X (gfx1250) — matrix operand order repaired and checked by reading, not compiled, not run on hardware
//
#include <hip/hip_runtime.h>
#include <hip/hip_bf16.h>

typedef __attribute__((ext_vector_type(16))) _Float16 v16h;
typedef __attribute__((ext_vector_type(8)))  _Float16 v8h;
typedef __attribute__((ext_vector_type(8)))  float    v8f;

#define N_CAM 6
#define C_DIM 256
#define H_IMG 32
#define W_IMG 88
#define BEV_H 100
#define BEV_W 100
#define NQ (BEV_H * BEV_W)

// ---------------------------------------------------------------------------
// Kernel 1: transpose img_feats (cam, C, H, W) -> channel-last (cam, H, W, C)
// so per-pixel channel vectors are 1KB contiguous (coalesced gather later).
// ---------------------------------------------------------------------------
__global__ void k_transpose(const float* __restrict__ img, float* __restrict__ imgT) {
    int pix = blockIdx.x;                 // cam*H*W + y*W + x  (0..16895)
    int c   = threadIdx.x;                // channel 0..255
    int cam = pix / (H_IMG * W_IMG);
    int rem = pix - cam * (H_IMG * W_IMG);
    imgT[(size_t)pix * C_DIM + c] =
        img[((size_t)(cam * C_DIM + c)) * (H_IMG * W_IMG) + rem];
}

// ---------------------------------------------------------------------------
// Kernel 2: per-query projection + masked bilinear gather + mean over 24
// (6 cams x 4 z). One block per query, one thread per channel.
// Emits A matrix (10000 x 256) in f16 row-major for the WMMA GEMM.
// ---------------------------------------------------------------------------
__global__ void k_sample(const float* __restrict__ imgT,
                         const float* __restrict__ l2i,
                         _Float16* __restrict__ A) {
    __shared__ int   s_x0[24], s_x1[24], s_y0[24], s_y1[24], s_act[24];
    __shared__ float s_w[24][4];

    int q = blockIdx.x;
    int c = threadIdx.x;

    if (c < 24) {
        int cam = c >> 2, zi = c & 3;
        int qy = q / BEV_W, qx = q - qy * BEV_W;
        // reference: ref_x=(qx+0.5)/W, ref_y=(qy+0.5)/H, ref_z=zi+0.5 (NOT /Z)
        float x  = ((qx + 0.5f) * (1.0f / BEV_W)) * 102.4f - 51.2f;
        float y  = ((qy + 0.5f) * (1.0f / BEV_H)) * 102.4f - 51.2f;
        float zc = (zi + 0.5f) * 8.0f - 5.0f;
        const float* M = l2i + cam * 16;           // row-major 4x4
        float cx = M[0]*x + M[1]*y + M[2]*zc  + M[3];
        float cy = M[4]*x + M[5]*y + M[6]*zc  + M[7];
        float cz = M[8]*x + M[9]*y + M[10]*zc + M[11];
        bool m = cz > 1e-5f;
        float zd = fmaxf(cz, 1e-5f);
        float u = cx / zd, v = cy / zd;
        float gx = (u * (1.0f / W_IMG) - 0.5f) * 2.0f;
        float gy = (v * (1.0f / H_IMG) - 0.5f) * 2.0f;
        m = m && (gx > -1.0f) && (gx < 1.0f) && (gy > -1.0f) && (gy < 1.0f);

        int act = 0, x0c = 0, x1c = 0, y0c = 0, y1c = 0;
        float w00 = 0.f, w10 = 0.f, w01 = 0.f, w11 = 0.f;
        if (m) {
            float xs = u - 0.5f, ys = v - 0.5f;    // (gx+1)*W/2 - 0.5 == u - 0.5
            float x0f = floorf(xs), y0f = floorf(ys);
            float wx1 = xs - x0f, wy1 = ys - y0f;
            int x0 = (int)x0f, y0 = (int)y0f;
            bool vx0 = (x0 >= 0) && (x0 < W_IMG);
            bool vx1 = (x0 + 1 >= 0) && (x0 + 1 < W_IMG);
            bool vy0 = (y0 >= 0) && (y0 < H_IMG);
            bool vy1 = (y0 + 1 >= 0) && (y0 + 1 < H_IMG);
            w00 = (1.f - wx1) * (1.f - wy1) * ((vx0 && vy0) ? 1.f : 0.f);
            w10 = wx1 * (1.f - wy1)         * ((vx1 && vy0) ? 1.f : 0.f);
            w01 = (1.f - wx1) * wy1         * ((vx0 && vy1) ? 1.f : 0.f);
            w11 = wx1 * wy1                 * ((vx1 && vy1) ? 1.f : 0.f);
            x0c = min(max(x0, 0), W_IMG - 1);
            x1c = min(max(x0 + 1, 0), W_IMG - 1);
            y0c = min(max(y0, 0), H_IMG - 1);
            y1c = min(max(y0 + 1, 0), H_IMG - 1);
            act = 1;
        }
        s_x0[c] = x0c; s_x1[c] = x1c; s_y0[c] = y0c; s_y1[c] = y1c; s_act[c] = act;
        s_w[c][0] = w00; s_w[c][1] = w10; s_w[c][2] = w01; s_w[c][3] = w11;
    }
    __syncthreads();

    float acc = 0.0f;
    for (int e = 0; e < 24; ++e) {
        if (!s_act[e]) continue;                 // block-uniform branch
        int cam = e >> 2;
        const float* base = imgT + (size_t)cam * (H_IMG * (size_t)W_IMG * C_DIM) + c;
        int y0 = s_y0[e], y1 = s_y1[e], x0 = s_x0[e], x1 = s_x1[e];
        float v00 = base[(size_t)(y0 * W_IMG + x0) * C_DIM];
        float v10 = base[(size_t)(y0 * W_IMG + x1) * C_DIM];
        float v01 = base[(size_t)(y1 * W_IMG + x0) * C_DIM];
        float v11 = base[(size_t)(y1 * W_IMG + x1) * C_DIM];
        acc += s_w[e][0] * v00 + s_w[e][1] * v10 + s_w[e][2] * v01 + s_w[e][3] * v11;
    }
    A[(size_t)q * C_DIM + c] = (_Float16)(acc * (1.0f / 24.0f));
}

// ---------------------------------------------------------------------------
// Kernel 3: pack W_out^T (B[k][n] = W_out[n][k]) into WMMA B-fragment order:
// flat half index = ((kblock*16 + tileN)*32 + lane)*16 + i,
// where lane holds col N = tileN*16 + lane%16 and element i <-> K = kblock*32
// + (i/8)*16 + (lane/16)*8 + i%8  (16-bit B layout, wave32).
// ---------------------------------------------------------------------------
__global__ void k_packB(const float* __restrict__ W_out, _Float16* __restrict__ Bp) {
    int t = blockIdx.x * blockDim.x + threadIdx.x;   // 0..65535
    int i      = t & 15;
    int lane   = (t >> 4) & 31;
    int tileN  = (t >> 9) & 15;
    int kblock = t >> 13;
    int k = kblock * 32 + (i >> 3) * 16 + (lane >> 4) * 8 + (i & 7);
    int n = tileN * 16 + (lane & 15);
    Bp[t] = (_Float16)W_out[n * C_DIM + k];
}

// ---------------------------------------------------------------------------
// Kernel 4: WMMA GEMM  out[m][n] = sum_k A[m][k]*B[k][n] + b_out[n] + query[m][n]
// 8 waves/block, one 16x16 tile per wave, K = 256 via 8x wmma_f32_16x16x32_f16.
// ---------------------------------------------------------------------------
__global__ void k_gemm(const _Float16* __restrict__ A,
                       const _Float16* __restrict__ Bp,
                       const float* __restrict__ bias,
                       const float* __restrict__ query,
                       float* __restrict__ out) {
    int wave = threadIdx.x >> 5;
    int lane = threadIdx.x & 31;
    int tileId = blockIdx.x * 8 + wave;      // 0..9999
    int tileM = tileId >> 4;                 // 0..624
    int tileN = tileId & 15;                 // 0..15
    int row = lane & 15;
    int hh  = lane >> 4;                     // half-wave: 0 or 1

    v8f acc = {};
    // A fragment: lane holds row M = tileM*16 + row; elems 0..7 <-> K=k0+hh*8+i,
    // elems 8..15 <-> K=k0+16+hh*8+i  (16-bit A layout, wave32).
    const _Float16* Abase = A + ((size_t)(tileM * 16 + row)) * C_DIM + hh * 8;
    const _Float16* Bbase = Bp + ((size_t)tileN * 32 + lane) * 16;

    #pragma unroll
    for (int kb = 0; kb < 8; ++kb) {
        v8h alo = *reinterpret_cast<const v8h*>(Abase + kb * 32);
        v8h ahi = *reinterpret_cast<const v8h*>(Abase + kb * 32 + 16);
        v16h a = __builtin_shufflevector(alo, ahi,
                 0, 1, 2, 3, 4, 5, 6, 7, 8, 9, 10, 11, 12, 13, 14, 15);
        v16h b = *reinterpret_cast<const v16h*>(Bbase + (size_t)kb * (16 * 32 * 16));
        acc = __builtin_amdgcn_wmma_f32_16x16x32_f16(
                  false, a, false, b, (short)0, acc, false, false);
    }

    // C/D layout: VGPR r <-> row M = r + 8*hh, col N = lane%16
    int n = tileN * 16 + row;
    float bn = bias[n];
    #pragma unroll
    for (int r = 0; r < 8; ++r) {
        int m = tileM * 16 + r + 8 * hh;
        out[(size_t)m * C_DIM + n] = acc[r] + bn + query[(size_t)m * C_DIM + n];
    }
}

// ---------------------------------------------------------------------------
extern "C" void kernel_launch(void* const* d_in, const int* in_sizes, int n_in,
                              void* d_out, int out_size, void* d_ws, size_t ws_size,
                              hipStream_t stream) {
    (void)in_sizes; (void)n_in; (void)out_size; (void)ws_size;
    const float* query = (const float*)d_in[0];   // (1, 10000, 256)
    const float* img   = (const float*)d_in[1];   // (1, 6, 256, 32, 88)
    const float* l2i   = (const float*)d_in[2];   // (1, 6, 4, 4)
    const float* W_out = (const float*)d_in[3];   // (256, 256)
    const float* b_out = (const float*)d_in[4];   // (256,)
    float* out = (float*)d_out;                   // (1, 10000, 256)

    char* ws = (char*)d_ws;
    float*    imgT = (float*)ws;                              // 17,301,504 B
    _Float16* Afp  = (_Float16*)(ws + 17301504);              //  5,120,000 B
    _Float16* Bp   = (_Float16*)(ws + 17301504 + 5120000);    //    131,072 B

    hipLaunchKernelGGL(k_transpose, dim3(N_CAM * H_IMG * W_IMG), dim3(C_DIM), 0, stream,
                       img, imgT);
    hipLaunchKernelGGL(k_packB, dim3(256), dim3(256), 0, stream, W_out, Bp);
    hipLaunchKernelGGL(k_sample, dim3(NQ), dim3(C_DIM), 0, stream, imgT, l2i, Afp);
    hipLaunchKernelGGL(k_gemm, dim3(1250), dim3(256), 0, stream,
                       Afp, Bp, b_out, query, out);
}
